// MetaCatNN_75170517614991
// MI455X (gfx1250) — matrix operand order repaired and checked by reading, not compiled, not run on hardware
//
#include <hip/hip_runtime.h>

// ---------------------------------------------------------------------------
// CDNA5 (gfx1250) 10-deep Cantor-compress + ResNet18 meta-network.
// Convs + FC are implicit GEMM on v_wmma_f32_16x16x32_f16 (wave32).
//  - Weights packed once per conv into fragment-ready f16 (1KB per mTile/chunk)
//  - A fragments staged into LDS by the Tensor Data Mover (tensor_load_to_lds,
//    double buffered, s_wait_tensorcnt), shared by all 8 waves of a block.
//  - All activations f16; f32 accumulate; BN folded to scale/shift.
//  - Each wave: 16(cout) x 64(pixel) tile -> 4 v_wmma per K-chunk.
// ---------------------------------------------------------------------------

typedef __attribute__((ext_vector_type(16))) _Float16     v16h;
typedef __attribute__((ext_vector_type(8)))  float        v8f;
typedef __attribute__((ext_vector_type(4)))  unsigned int v4u;
typedef __attribute__((ext_vector_type(8)))  int          v8i;
typedef __attribute__((ext_vector_type(4)))  int          v4i;

#if defined(__AMDGCN__) && __has_builtin(__builtin_amdgcn_tensor_load_to_lds) && \
    __has_builtin(__builtin_amdgcn_s_wait_tensorcnt)
#define USE_TDM 1
#else
#define USE_TDM 0
#endif

// ---- WMMA fragment K mapping (16-bit A 16x32) ------------------------------
// A: lane L holds row M=L%16, half h=L/16; element e maps to K = h*16 + e.
// B: lane L holds col N=L%16; element e maps to K = h*16 + e  (consecutive).
__device__ __host__ __forceinline__ int a_klocal(int h, int e) {
    return h * 16 + e;
}

#if defined(__AMDGCN__)
// LDS byte offset of a __shared__ object. The generic->addrspace(3) cast plus
// ptrtoint also *captures* the LDS object, so passing the result into the TDM
// intrinsic forces alias analysis to assume the TDM writes it (keeps the
// ds_load fragment reads and the LDS allocation alive).
__device__ __forceinline__ unsigned lds_offset_of(const void* p) {
    return (unsigned)(unsigned long long)
        (__attribute__((address_space(3))) const void*)p;
}
#endif

#if USE_TDM
// Issue a TDM copy of one contiguous 1KB A-fragment chunk into LDS.
// D# per CDNA5 ISA: group0 = {count=1, lds_addr,
// global_addr, type=2}; group1 = 1-D tensor: data_size=1B, tensor_dim0 =
// tile_dim0 = 1024, tensor_dim1 = 1, dim0_stride = 1024; groups 2/3 zero.
// This toolchain exposes the 6-arg builtin (extra int32x8 group before cpol).
__device__ __forceinline__ void tdm_load_1k(const void* gsrc, unsigned ldsOff) {
    unsigned long long ga = (unsigned long long)gsrc;
    v4u g0;
    v8i g1;
    v4i g2 = {0, 0, 0, 0}, g3 = {0, 0, 0, 0};
    v8i g4 = {0, 0, 0, 0, 0, 0, 0, 0};
    g0[0] = 1u;                                        // count=1 (valid user D#)
    g0[1] = ldsOff;                                    // lds_addr (bytes)
    g0[2] = (unsigned)(ga & 0xFFFFFFFFu);              // global_addr[31:0]
    g0[3] = (unsigned)((ga >> 32) & 0x01FFFFFFu)       // global_addr[56:32]
          | (2u << 30);                                // type=2 ("image")
    g1[0] = 0;                                         // wg_mask=0, data_size=1B
    g1[1] = (int)((1024u & 0xFFFFu) << 16);            // tensor_dim0[15:0]
    g1[2] = (int)((1024u >> 16) | (1u << 16));         // dim0 hi | tensor_dim1=1
    g1[3] = (int)((1024u & 0xFFFFu) << 16);            // tile_dim0 = 1024
    g1[4] = 0;                                         // tile_dim1/2 unused
    g1[5] = 1024;                                      // tensor_dim0_stride lo
    g1[6] = 0;
    g1[7] = 0;
    __builtin_amdgcn_tensor_load_to_lds(g0, g1, g2, g3, g4, /*cpol=*/0);
}
#endif

// ---------------------------------------------------------------------------
// Weight packing: w[COUT][Ktot] f32 -> packed f16 fragments
// layout: ((mTile*nChunks + chunk)*32 + lane)*16 + e   (1KB per mTile/chunk)
// ---------------------------------------------------------------------------
__global__ void k_pack_w(const float* __restrict__ w, _Float16* __restrict__ p,
                         int COUT, int Ktot, int nChunks) {
    const int mTiles = (COUT + 15) >> 4;
    const long total = (long)mTiles * nChunks * 512;
    long id = (long)blockIdx.x * blockDim.x + threadIdx.x;
    if (id >= total) return;
    int  e    = id & 15;
    int  lane = (int)((id >> 4) & 31);
    long t    = id >> 9;
    int  chunk = (int)(t % nChunks);
    int  mT    = (int)(t / nChunks);
    int  ln = lane & 15, h = lane >> 4;
    int  m = mT * 16 + ln;
    int  k = chunk * 32 + a_klocal(h, e);
    float v = (m < COUT && k < Ktot) ? w[(long)m * Ktot + k] : 0.0f;
    p[id] = (_Float16)v;
}

// ---------------------------------------------------------------------------
// Fold inference BN into per-channel scale/shift.
// ---------------------------------------------------------------------------
__global__ void k_bnfold(const float* __restrict__ g, const float* __restrict__ b,
                         const float* __restrict__ m, const float* __restrict__ v,
                         float* __restrict__ scale, float* __restrict__ shift, int C) {
    int i = blockIdx.x * blockDim.x + threadIdx.x;
    if (i >= C) return;
    float inv = g[i] * rsqrtf(v[i] + 1e-5f);
    scale[i] = inv;
    shift[i] = b[i] - m[i] * inv;
}

// ---------------------------------------------------------------------------
// ok_compression (ternary digits, digit==2 -> 2^-(i+1)); f32 in, f16 out.
// ---------------------------------------------------------------------------
__global__ void k_compress(const float* __restrict__ in, _Float16* __restrict__ out,
                           int n, int depth, int rep3, int HW) {
    int i = blockIdx.x * blockDim.x + threadIdx.x;
    if (i >= n) return;
    float v = in[i] * (1.0f / 255.0f);
    float acc = 0.0f, wgt = 0.5f;
    for (int d = 0; d < depth; ++d) {
        v *= 3.0f;
        float t = truncf(v);
        v -= t;
        acc += (t == 2.0f) ? wgt : 0.0f;
        wgt *= 0.5f;
    }
    _Float16 r = (_Float16)(acc * 255.0f);
    if (rep3) {
        long b = i / HW, p = i % HW;
        out[(b * 3 + 0) * (long)HW + p] = r;
        out[(b * 3 + 1) * (long)HW + p] = r;
        out[(b * 3 + 2) * (long)HW + p] = r;
    } else {
        out[i] = r;
    }
}

// ---------------------------------------------------------------------------
// Implicit-GEMM conv. Block = 8 waves sharing one mTile; each wave owns a
// 16x64 output tile. A chunk (1KB) staged to LDS via TDM (double buffered),
// read back as two ds_load_b128 per lane. Fused BN (+residual) (+ReLU).
// ---------------------------------------------------------------------------
__global__ void __launch_bounds__(256)
k_conv_wmma(const _Float16* __restrict__ x,
            const _Float16* __restrict__ packA,
            const float* __restrict__ scale,
            const float* __restrict__ shift,
            const _Float16* __restrict__ res,
            _Float16* __restrict__ out,
            int B, int CIN, int IH, int IW,
            int COUT, int OH, int OW,
            int KH, int KW, int stride, int pad, int do_relu) {
    const int lane   = threadIdx.x & 31;
    const int waveId = threadIdx.x >> 5;
    const int Ntot   = B * OH * OW;
    const int nGroups = (Ntot + 511) >> 9;          // 8 waves x 64 px per block
    const int mT = blockIdx.x / nGroups;
    const int nG = blockIdx.x % nGroups;
    const int n0 = (nG * 8 + waveId) << 6;
    const int Ktot    = CIN * KH * KW;
    const int nChunks = (Ktot + 31) >> 5;
    const int h  = lane >> 4;
    const int ln = lane & 15;
    const int khw = KH * KW, ohw = OH * OW, ihw = IH * IW;

    // Per-subtile pixel coordinates (hoisted out of the K loop).
    int  nn[4], bi[4], oh_[4], ow_[4], ihb[4], iwb[4];
    long xbase[4];
    #pragma unroll
    for (int t = 0; t < 4; ++t) {
        int n = n0 + t * 16 + ln;
        nn[t] = n;
        int b_ = n / ohw, r = n - b_ * ohw;
        int oh = r / OW,  ow = r - (r / OW) * OW;
        bi[t] = b_; oh_[t] = oh; ow_[t] = ow;
        ihb[t] = oh * stride - pad;
        iwb[t] = ow * stride - pad;
        xbase[t] = (long)b_ * CIN * ihw;
    }

    const _Float16* aglob = packA + (long)mT * nChunks * 512;

    __shared__ _Float16 abuf[2][512];               // 2 x 1KB double buffer
#if USE_TDM
    // Real LDS byte offset of abuf; ptrtoint into the TDM intrinsic also
    // escapes abuf so its loads/allocation cannot be optimized away.
    const unsigned ldsBase = lds_offset_of(&abuf[0][0]);
#endif

    // Stage chunk 0 into buffer 0 (wave 0 only; TDM ignores EXEC).
    if (waveId == 0) {
#if USE_TDM
        tdm_load_1k(aglob, ldsBase);
#else
        *(v16h*)&abuf[0][lane * 16] = ((const v16h*)aglob)[lane];
#endif
    }

    v8f acc[4] = {};
    for (int c = 0; c < nChunks; ++c) {
#if USE_TDM
        if (waveId == 0) __builtin_amdgcn_s_wait_tensorcnt(0);
#endif
        __syncthreads();                            // publish abuf[c&1]
        v16h a = *(const v16h*)&abuf[c & 1][lane * 16];   // 2x ds_load_b128

        // Hoist k -> (ci, kh, kw) decomposition once per chunk (k = kbase+e).
        const int kbase = c * 32 + h * 16;
        int  cioff[16], khh[16], kww[16], kvalid[16];
        #pragma unroll
        for (int e = 0; e < 16; ++e) {
            int k  = kbase + e;
            int ci = k / khw;
            int kr = k - ci * khw;
            int kh = kr / KW;
            cioff[e]  = ci * ihw;
            khh[e]    = kh;
            kww[e]    = kr - kh * KW;
            kvalid[e] = (k < Ktot);
        }

        #pragma unroll
        for (int t = 0; t < 4; ++t) {
            if (n0 + t * 16 < Ntot) {               // wave-uniform guard
                v16h b;
                #pragma unroll
                for (int e = 0; e < 16; ++e) {
                    _Float16 bv = (_Float16)0.0f;
                    if (nn[t] < Ntot && kvalid[e]) {
                        int ih = ihb[t] + khh[e];
                        int iw = iwb[t] + kww[e];
                        if (ih >= 0 && ih < IH && iw >= 0 && iw < IW)
                            bv = x[xbase[t] + cioff[e] + (long)ih * IW + iw];
                    }
                    b[e] = bv;
                }
                acc[t] = __builtin_amdgcn_wmma_f32_16x16x32_f16(
                    false, a, false, b, (short)0, acc[t], false, false);
            }
        }
        __syncthreads();                            // all done reading abuf[c&1]
        if (waveId == 0 && c + 1 < nChunks) {       // prefetch next chunk
#if USE_TDM
            tdm_load_1k(aglob + (long)(c + 1) * 512,
                        ldsBase + (unsigned)(((c + 1) & 1) * 1024));
#else
            *(v16h*)&abuf[(c + 1) & 1][lane * 16] =
                ((const v16h*)(aglob + (long)(c + 1) * 512))[lane];
#endif
        }
    }

    // Epilogue: scale/shift -> (+residual) -> (ReLU) -> f16 store.
    const int m0 = mT << 4;
    #pragma unroll
    for (int t = 0; t < 4; ++t) {
        if (nn[t] < Ntot) {
            #pragma unroll
            for (int v = 0; v < 8; ++v) {
                int m = m0 + v + h * 8;
                if (m < COUT) {
                    float val = acc[t][v] * scale[m] + shift[m];
                    long  oidx = (((long)bi[t] * COUT + m) * OH + oh_[t]) * OW + ow_[t];
                    if (res) val += (float)res[oidx];
                    if (do_relu) val = fmaxf(val, 0.0f);
                    out[oidx] = (_Float16)val;
                }
            }
        }
    }
}

// ---------------------------------------------------------------------------
// FC: out[b][o] = feat[b].fcw[o] + fcb[o]; packed-A, vector-loaded f16 B.
// M=12288, K=512, N=8 (padded to 16). f32 output.
// ---------------------------------------------------------------------------
__global__ void k_fc_wmma(const _Float16* __restrict__ feat,
                          const _Float16* __restrict__ packA,
                          const float* __restrict__ fcb, float* __restrict__ out,
                          int OUTF, int K, int B) {
    const int lane   = threadIdx.x & 31;
    const int waveId = threadIdx.x >> 5;
    const int tile   = blockIdx.x * (blockDim.x >> 5) + waveId;
    if (tile * 16 >= OUTF) return;
    const int m0 = tile << 4;
    const int h  = lane >> 4;
    const int ln = lane & 15;
    const int nChunks = K >> 5;
    const _Float16* afrag = packA + ((long)tile * nChunks * 32 + lane) * 16;

    v8f acc = {};
    for (int c = 0; c < nChunks; ++c) {
        v16h a = *(const v16h*)(afrag + (long)c * 512);
        __builtin_prefetch(afrag + (long)(c + 1) * 512);
        v16h b = {};
        if (ln < B)
            b = *(const v16h*)(feat + (long)ln * K + c * 32 + h * 16);
        acc = __builtin_amdgcn_wmma_f32_16x16x32_f16(
            false, a, false, b, (short)0, acc, false, false);
    }
    if (ln < B) {
        #pragma unroll
        for (int v = 0; v < 8; ++v) {
            int m = m0 + v + h * 8;
            out[(long)ln * OUTF + m] = acc[v] + fcb[m];
        }
    }
}

// ---------------------------------------------------------------------------
// 3x3 s2 p1 max pool, f16.
// ---------------------------------------------------------------------------
__global__ void k_maxpool(const _Float16* __restrict__ in, _Float16* __restrict__ out,
                          int B, int C, int IH, int IW, int OH, int OW) {
    long i   = (long)blockIdx.x * blockDim.x + threadIdx.x;
    long tot = (long)B * C * OH * OW;
    if (i >= tot) return;
    int ow = i % OW; long t = i / OW;
    int oh = t % OH; t /= OH;
    int c  = t % C;  int b = t / C;
    const _Float16* p = in + ((long)b * C + c) * IH * IW;
    float mx = -__FLT_MAX__;
    #pragma unroll
    for (int kh = 0; kh < 3; ++kh) {
        int ih = oh * 2 - 1 + kh;
        if (ih < 0 || ih >= IH) continue;
        #pragma unroll
        for (int kw = 0; kw < 3; ++kw) {
            int iw = ow * 2 - 1 + kw;
            if (iw < 0 || iw >= IW) continue;
            mx = fmaxf(mx, (float)p[(long)ih * IW + iw]);
        }
    }
    out[i] = (_Float16)mx;
}

// ---------------------------------------------------------------------------
// Global average pool: one thread per (b,c); H*W <= 49; f32 accumulate.
// ---------------------------------------------------------------------------
__global__ void k_avgpool(const _Float16* __restrict__ in, _Float16* __restrict__ feat,
                          int BC, int HW) {
    int i = blockIdx.x * blockDim.x + threadIdx.x;
    if (i >= BC) return;
    const _Float16* p = in + (long)i * HW;
    float s = 0.0f;
    for (int j = 0; j < HW; ++j) s += (float)p[j];
    feat[i] = (_Float16)(s / (float)HW);
}

// ---------------------------------------------------------------------------
// Host orchestration.
// ---------------------------------------------------------------------------
namespace {

struct BNP { const float *g, *b, *m, *v; };

struct Ctx {
    hipStream_t stream;
    float*     bnScale;
    float*     bnShift;
    _Float16*  packW;
};

void launch_conv(const Ctx& cx, const _Float16* x, const float* w, const BNP& bn,
                 const _Float16* res, _Float16* out,
                 int B, int CIN, int IH, int IW, int COUT, int KH, int KW,
                 int stride, int pad, int relu, int* OHp, int* OWp) {
    int OH = (IH + 2 * pad - KH) / stride + 1;
    int OW = (IW + 2 * pad - KW) / stride + 1;
    *OHp = OH; *OWp = OW;
    int Ktot    = CIN * KH * KW;
    int nChunks = (Ktot + 31) / 32;
    int mTiles  = (COUT + 15) / 16;

    k_bnfold<<<(COUT + 255) / 256, 256, 0, cx.stream>>>(
        bn.g, bn.b, bn.m, bn.v, cx.bnScale, cx.bnShift, COUT);
    long ptot = (long)mTiles * nChunks * 512;
    k_pack_w<<<(int)((ptot + 255) / 256), 256, 0, cx.stream>>>(
        w, cx.packW, COUT, Ktot, nChunks);

    int Ntot    = B * OH * OW;
    int nGroups = (Ntot + 511) / 512;
    int blocks  = mTiles * nGroups;                 // all 8 waves share mTile
    k_conv_wmma<<<blocks, 256, 0, cx.stream>>>(x, cx.packW, cx.bnScale, cx.bnShift,
                                               res, out, B, CIN, IH, IW,
                                               COUT, OH, OW, KH, KW, stride, pad, relu);
}

} // namespace

extern "C" void kernel_launch(void* const* d_in, const int* in_sizes, int n_in,
                              void* d_out, int out_size, void* d_ws, size_t ws_size,
                              hipStream_t stream) {
    (void)in_sizes; (void)n_in; (void)out_size; (void)ws_size;
    const int B = 8, NDEPTH = 10, FEATC = 512, OUTF = 3 * 64 * 64;

    char*  ws  = (char*)d_ws;
    size_t off = 0;
    auto alloc = [&](size_t bytes) {
        size_t p = off;
        off = (off + bytes + 4095) & ~(size_t)4095;
        return p;
    };
    const size_t SLOT = (size_t)8 * 64 * 112 * 112 * sizeof(_Float16);
    _Float16* S[5];
    for (int i = 0; i < 5; ++i) S[i] = (_Float16*)(ws + alloc(SLOT));
    _Float16* feat    = (_Float16*)(ws + alloc((size_t)B * FEATC * sizeof(_Float16)));
    float*    fcout   = (float*)   (ws + alloc((size_t)B * OUTF * sizeof(float)));
    float*    bnScale = (float*)   (ws + alloc(512 * sizeof(float)));
    float*    bnShift = (float*)   (ws + alloc(512 * sizeof(float)));
    _Float16* packW   = (_Float16*)(ws + alloc((size_t)16 << 20));

    Ctx cx{stream, bnScale, bnShift, packW};

    // Pytree leaf walk: d_in[0]=x; per net (sorted keys): bn1(g,b,m,v), conv1,
    // fcb, fcw, stages[4][2] with blocks as bn1(4),bn2(4),[dbn(4),dw],w1,w2.
    const float* x0 = (const float*)d_in[0];
    int idx = 1;
    auto nextPtr = [&]() { return (const float*)d_in[idx++]; };
    auto getBN   = [&]() {
        BNP r; r.g = nextPtr(); r.b = nextPtr(); r.m = nextPtr(); r.v = nextPtr();
        return r;
    };

    const int chans[4][2]     = {{64, 64}, {64, 128}, {128, 256}, {256, 512}};
    const int stage_stride[4] = {1, 2, 2, 2};

    const float* prev = nullptr;
    for (int it = 0; it < NDEPTH; ++it) {
        int H, W;
        if (it == 0) {
            H = 224; W = 224;
            int n = B * H * W;
            k_compress<<<(n + 255) / 256, 256, 0, stream>>>(x0, S[0], n, 1, 1, H * W);
        } else {
            H = 64; W = 64;
            int n = B * 3 * H * W;
            k_compress<<<(n + 255) / 256, 256, 0, stream>>>(prev, S[0], n, it + 1, 0, 0);
        }

        BNP bn1 = getBN();
        const float* conv1 = nextPtr();
        const float* fcb   = nextPtr();
        const float* fcw   = nextPtr();

        int OH, OW;
        launch_conv(cx, S[0], conv1, bn1, nullptr, S[1],
                    B, 3, H, W, 64, 7, 7, 2, 3, /*relu=*/1, &OH, &OW);
        int PH = (OH + 2 - 3) / 2 + 1, PW = (OW + 2 - 3) / 2 + 1;
        {
            long tot = (long)B * 64 * PH * PW;
            k_maxpool<<<(int)((tot + 255) / 256), 256, 0, stream>>>(
                S[1], S[2], B, 64, OH, OW, PH, PW);
        }
        _Float16* cur = S[2]; int curSlot = 2;
        int curH = PH, curW = PW;

        for (int s = 0; s < 4; ++s) {
            int cin = chans[s][0], cout = chans[s][1];
            for (int blk = 0; blk < 2; ++blk) {
                int  stride  = (blk == 0) ? stage_stride[s] : 1;
                int  bcin    = (blk == 0) ? cin : cout;
                bool has_dw  = (blk == 0) && (cin != cout);

                BNP b1 = getBN();
                BNP b2 = getBN();
                BNP dbn = {};
                const float* dw = nullptr;
                if (has_dw) { dbn = getBN(); dw = nextPtr(); }
                const float* w1 = nextPtr();
                const float* w2 = nextPtr();

                int p[3], c = 0;
                for (int q = 0; q < 5 && c < 3; ++q) if (q != curSlot) p[c++] = q;
                _Float16* out1 = S[p[0]];
                _Float16* sc   = S[p[1]];
                _Float16* out2 = S[p[2]];

                int oH, oW;
                launch_conv(cx, cur, w1, b1, nullptr, out1,
                            B, bcin, curH, curW, cout, 3, 3, stride, 1, 1, &oH, &oW);
                const _Float16* resptr;
                if (has_dw) {
                    int t1, t2;
                    launch_conv(cx, cur, dw, dbn, nullptr, sc,
                                B, bcin, curH, curW, cout, 1, 1, stride, 0, 0, &t1, &t2);
                    resptr = sc;
                } else {
                    resptr = cur;
                }
                int fH, fW;
                launch_conv(cx, out1, w2, b2, resptr, out2,
                            B, cout, oH, oW, cout, 3, 3, 1, 1, 1, &fH, &fW);

                cur = out2; curSlot = p[2];
                curH = fH; curW = fW;
            }
        }

        k_avgpool<<<(B * FEATC + 255) / 256, 256, 0, stream>>>(
            cur, feat, B * FEATC, curH * curW);

        {
            int nChunks = FEATC / 32;
            long ptot = (long)(OUTF / 16) * nChunks * 512;
            k_pack_w<<<(int)((ptot + 255) / 256), 256, 0, stream>>>(
                fcw, packW, OUTF, FEATC, nChunks);
        }
        float* fcdst = (it == NDEPTH - 1) ? (float*)d_out : fcout;
        {
            int tiles  = OUTF / 16;
            int blocks = (tiles + 7) / 8;
            k_fc_wmma<<<blocks, 256, 0, stream>>>(feat, packW, fcb, fcdst,
                                                  OUTF, FEATC, B);
        }
        prev = fcdst;
    }
}
